// MaskablePPOPolicy_CONCAT_82128364634635
// MI455X (gfx1250) — compile-verified, matrix operand order checked by reading, not run on hardware
//
#include <hip/hip_runtime.h>
#include <hip/hip_bf16.h>
#include <math.h>

// ---------------- problem constants ----------------
#define E       128
#define HID     256           // 2*E
#define SEQ     128           // S
#define MAXN    33
#define BSZ     64
#define NN      (BSZ*MAXN)    // 2112 nodes
#define FDIM    (E+3)         // 131
#define ROWS    (SEQ*NN)      // 270336
#define GATES4  (4*HID)       // 1024
#define KCAT    (2*HID)       // 512

typedef __attribute__((ext_vector_type(16))) __bf16 v16bf;
typedef __attribute__((ext_vector_type(8)))  float  v8f;

// ---------------- workspace layout (bytes, all 256-aligned) ----------------
// muB  : ROWS*128 bf16            =  69,206,016
// mpB  : SEQ*BSZ*128 bf16         =   2,097,152
// x    : ROWS*256 bf16            = 138,412,032
// Wcat : 1024*512 bf16            =   1,048,576
// W6B  : 128*128 bf16             =      32,768
// W7B  : 128*128 bf16             =      32,768
// bcat : 1024 f32                 =       4,096
// gates: 2112*1024 f32            =   8,650,752
// hB   : 2112*256 bf16            =   1,081,344
// c    : 2112*256 f32             =   2,162,688
static constexpr size_t MUB_OFF  = 0;
static constexpr size_t MPB_OFF  = 69206016;
static constexpr size_t X_OFF    = 71303168;
static constexpr size_t WCAT_OFF = 209715200;
static constexpr size_t W6B_OFF  = 210763776;
static constexpr size_t W7B_OFF  = 210796544;
static constexpr size_t BCAT_OFF = 210829312;
static constexpr size_t GATE_OFF = 210833408;
static constexpr size_t HB_OFF   = 219484160;
static constexpr size_t C_OFF    = 220565504;

// ---------------- WMMA bf16 fragment loader ----------------
// 16-bit A (16x32) / B (32x16) striping per CDNA5 ISA 7.12.2:
// lane L covers row/col = L%16; k-chunks {base..base+7} and {base+16..base+23},
// base = (L<16 ? 0 : 8).  Two 16B vector loads -> global_load_b128.
__device__ __forceinline__ v16bf load_frag(const __bf16* rowptr, unsigned lane) {
    const int base = (lane & 16u) ? 8 : 0;
    union { v16bf v; uint4 u[2]; } f;
    f.u[0] = *reinterpret_cast<const uint4*>(rowptr + base);
    f.u[1] = *reinterpret_cast<const uint4*>(rowptr + base + 16);
    return f.v;
}

__device__ __forceinline__ v8f wmma_bf16(v16bf a, v16bf b, v8f c) {
    return __builtin_amdgcn_wmma_f32_16x16x32_bf16(
        false, a, false, b, (short)0, c, false, false);
}

// ---------------- prologue: weight packing + state init ----------------
__global__ __launch_bounds__(256) void prep_kernel(
    const float* __restrict__ W6, const float* __restrict__ W7,
    const float* __restrict__ W_ih, const float* __restrict__ W_hh,
    const float* __restrict__ b_ih, const float* __restrict__ b_hh,
    __bf16* __restrict__ W6B, __bf16* __restrict__ W7B,
    __bf16* __restrict__ WcatB, float* __restrict__ bcat,
    __bf16* __restrict__ hB, float* __restrict__ cbuf)
{
    const unsigned tid = blockIdx.x * 256u + threadIdx.x;   // grid = 540672
    if (tid < 1024u * 512u) {
        const unsigned j = tid >> 9, k = tid & 511u;
        const float w = (k < 256u) ? W_ih[j*256u + k] : W_hh[j*256u + (k - 256u)];
        WcatB[tid] = (__bf16)w;
    }
    if (tid < 128u * 128u) {
        W6B[tid] = (__bf16)W6[tid];
        W7B[tid] = (__bf16)W7[tid];
    }
    if (tid < 1024u) bcat[tid] = b_ih[tid] + b_hh[tid];
    if (tid < (unsigned)(NN * HID)) {
        hB[tid]   = (__bf16)0.0f;
        cbuf[tid] = 0.0f;
    }
}

// ---------------- per-batch mean pool (segments are exactly 33 nodes) ------
__global__ __launch_bounds__(256) void meanpool_kernel(
    const float* __restrict__ features, __bf16* __restrict__ mpB)
{
    const unsigned tid = blockIdx.x * 256u + threadIdx.x;   // grid = S*B*128
    if (tid >= (unsigned)(SEQ * BSZ * E)) return;
    const unsigned s = tid >> 13;            // /(B*128)
    const unsigned rem = tid & 8191u;
    const unsigned b = rem >> 7, e = rem & 127u;
    float sum = 0.f;
    const size_t base = ((size_t)s * NN + (size_t)b * MAXN) * FDIM + e;
    #pragma unroll 3
    for (int j = 0; j < MAXN; ++j) sum += features[base + (size_t)j * FDIM];
    mpB[tid] = (__bf16)(sum * (1.0f / (float)MAXN));
}

// ---------------- strip mu out of features, convert to bf16 ----------------
__global__ __launch_bounds__(256) void muconv_kernel(
    const float* __restrict__ features, __bf16* __restrict__ muB)
{
    const unsigned tid = blockIdx.x * 256u + threadIdx.x;   // grid = ROWS*128
    if (tid >= (unsigned)(ROWS * E)) return;
    const unsigned row = tid >> 7, e = tid & 127u;
    muB[tid] = (__bf16)features[(size_t)row * FDIM + e];
}

// ---------------- x = relu([meanpool@W6^T + b6, mu@W7^T + b7]) -------------
// wave task: (Mtile of 16 rows, half in {g,l}, 64-col group); K = 128.
__global__ __launch_bounds__(256) void xgemm_kernel(
    const __bf16* __restrict__ muB, const __bf16* __restrict__ mpB,
    const __bf16* __restrict__ W6B, const __bf16* __restrict__ W7B,
    const float* __restrict__ b6, const float* __restrict__ b7,
    const float* __restrict__ features, __bf16* __restrict__ x)
{
    const unsigned wave = threadIdx.x >> 5, lane = threadIdx.x & 31u;
    const unsigned task = blockIdx.x * 8u + wave;           // 67584 tasks
    if (task >= (unsigned)(ROWS / 16) * 4u) return;
    const unsigned mtile = task >> 2;
    const unsigned hf    = (task >> 1) & 1u;                // 0: g-half, 1: l-half
    const unsigned cg    = task & 1u;                       // 64-col group

    // per-lane A row pointer (row M = lane%16 of this tile)
    const unsigned r  = mtile * 16u + (lane & 15u);
    const unsigned s_ = r / NN, n_ = r - s_ * NN;
    const __bf16* arow;
    if (hf == 0) {
        const int b = (int)(features[(size_t)n_ * FDIM + E] + 0.5f); // batch id
        arow = mpB + ((size_t)s_ * BSZ + (size_t)b) * E;
    } else {
        arow = muB + (size_t)r * E;
    }
    const __bf16* wb = hf ? W7B : W6B;
    const float*  bb = hf ? b7  : b6;

    v8f acc0 = {}, acc1 = {}, acc2 = {}, acc3 = {};
    #pragma unroll
    for (int kk = 0; kk < E; kk += 32) {
        const v16bf a = load_frag(arow + kk, lane);
        const unsigned c0 = cg * 64u + (lane & 15u);
        acc0 = wmma_bf16(a, load_frag(wb + (size_t)(c0 +  0u) * E + kk, lane), acc0);
        acc1 = wmma_bf16(a, load_frag(wb + (size_t)(c0 + 16u) * E + kk, lane), acc1);
        acc2 = wmma_bf16(a, load_frag(wb + (size_t)(c0 + 32u) * E + kk, lane), acc2);
        acc3 = wmma_bf16(a, load_frag(wb + (size_t)(c0 + 48u) * E + kk, lane), acc3);
    }
    // store: lane L -> col = base + L%16 ; vgpr i -> row mtile*16 + i + (L>=16)*8
    const unsigned rbase = mtile * 16u + ((lane >> 4) << 3);
    v8f accs[4] = {acc0, acc1, acc2, acc3};
    #pragma unroll
    for (int nt = 0; nt < 4; ++nt) {
        const unsigned col = cg * 64u + nt * 16u + (lane & 15u);
        const float bias = bb[col];
        #pragma unroll
        for (int i = 0; i < 8; ++i) {
            const float v = fmaxf(accs[nt][i] + bias, 0.0f);
            x[(size_t)(rbase + i) * (2*E) + hf * E + col] = (__bf16)v;
        }
    }
}

// ---------------- per-step gates GEMM: [x_t | h] (2112x512) @ Wcat^T -------
__global__ __launch_bounds__(256) void gates_gemm_kernel(
    const __bf16* __restrict__ x, const __bf16* __restrict__ hB,
    const __bf16* __restrict__ WcatB, float* __restrict__ gates, int s)
{
    const unsigned wave = threadIdx.x >> 5, lane = threadIdx.x & 31u;
    const unsigned task = blockIdx.x * 8u + wave;            // 2112 tasks
    if (task >= (unsigned)(NN / 16) * 16u) return;
    const unsigned mtile = task >> 4;
    const unsigned ncol0 = (task & 15u) * 64u;

    const unsigned n_l = mtile * 16u + (lane & 15u);
    const __bf16* xr = x  + ((size_t)s * NN + n_l) * (2*E);
    const __bf16* hr = hB + (size_t)n_l * HID;

    v8f acc0 = {}, acc1 = {}, acc2 = {}, acc3 = {};
    #pragma unroll
    for (int kk = 0; kk < KCAT; kk += 32) {
        const __bf16* ap = (kk < 2*E) ? (xr + kk) : (hr + (kk - 2*E));
        const v16bf a = load_frag(ap, lane);
        const unsigned c0 = ncol0 + (lane & 15u);
        acc0 = wmma_bf16(a, load_frag(WcatB + (size_t)(c0 +  0u) * KCAT + kk, lane), acc0);
        acc1 = wmma_bf16(a, load_frag(WcatB + (size_t)(c0 + 16u) * KCAT + kk, lane), acc1);
        acc2 = wmma_bf16(a, load_frag(WcatB + (size_t)(c0 + 32u) * KCAT + kk, lane), acc2);
        acc3 = wmma_bf16(a, load_frag(WcatB + (size_t)(c0 + 48u) * KCAT + kk, lane), acc3);
    }
    const unsigned rbase = mtile * 16u + ((lane >> 4) << 3);
    v8f accs[4] = {acc0, acc1, acc2, acc3};
    #pragma unroll
    for (int nt = 0; nt < 4; ++nt) {
        const unsigned col = ncol0 + nt * 16u + (lane & 15u);
        #pragma unroll
        for (int i = 0; i < 8; ++i)
            gates[(size_t)(rbase + i) * GATES4 + col] = accs[nt][i];
    }
}

// ---------------- per-step LSTM pointwise + fused masked logit -------------
__global__ __launch_bounds__(256) void lstm_update_kernel(
    const float* __restrict__ gates, const float* __restrict__ bcat,
    float* __restrict__ cbuf, __bf16* __restrict__ hB,
    const float* __restrict__ W5, const float* __restrict__ b5,
    const float* __restrict__ features, float* __restrict__ out, int s)
{
    const unsigned lane = threadIdx.x & 31u;
    const unsigned node = blockIdx.x * 8u + (threadIdx.x >> 5); // 2112 nodes
    if (node >= NN) return;

    float lsum = 0.0f;
    #pragma unroll
    for (int t = 0; t < 8; ++t) {
        const unsigned j = lane + (t << 5);                  // 0..255
        const size_t gb = (size_t)node * GATES4 + j;
        const float gi = gates[gb          ] + bcat[j          ];
        const float gf = gates[gb + 1*HID  ] + bcat[j + 1*HID  ];
        const float gg = gates[gb + 2*HID  ] + bcat[j + 2*HID  ];
        const float go = gates[gb + 3*HID  ] + bcat[j + 3*HID  ];
        const float is = 1.0f / (1.0f + expf(-gi));
        const float fs = 1.0f / (1.0f + expf(-gf));
        const float os = 1.0f / (1.0f + expf(-go));
        const size_t hi = (size_t)node * HID + j;
        const float cn = fs * cbuf[hi] + is * tanhf(gg);
        const float h  = os * tanhf(cn);
        cbuf[hi] = cn;
        hB[hi]   = (__bf16)h;
        lsum += fmaxf(h, 0.0f) * W5[j];
    }
    #pragma unroll
    for (int off = 16; off > 0; off >>= 1)
        lsum += __shfl_xor(lsum, off, 32);
    if (lane == 0) {
        const bool reach = features[((size_t)s * NN + node) * FDIM + (E + 1)] > 0.5f;
        out[(size_t)s * NN + node] = reach ? (lsum + b5[0]) : -INFINITY;
    }
}

// ---------------- host orchestration ----------------
extern "C" void kernel_launch(void* const* d_in, const int* in_sizes, int n_in,
                              void* d_out, int out_size, void* d_ws, size_t ws_size,
                              hipStream_t stream) {
    const float* features = (const float*)d_in[0];
    const float* W6   = (const float*)d_in[1];
    const float* b6   = (const float*)d_in[2];
    const float* W7   = (const float*)d_in[3];
    const float* b7   = (const float*)d_in[4];
    const float* W5   = (const float*)d_in[5];
    const float* b5   = (const float*)d_in[6];
    const float* W_ih = (const float*)d_in[7];
    const float* W_hh = (const float*)d_in[8];
    const float* b_ih = (const float*)d_in[9];
    const float* b_hh = (const float*)d_in[10];
    float* out = (float*)d_out;

    char* ws = (char*)d_ws;
    __bf16* muB   = (__bf16*)(ws + MUB_OFF);
    __bf16* mpB   = (__bf16*)(ws + MPB_OFF);
    __bf16* xbuf  = (__bf16*)(ws + X_OFF);
    __bf16* WcatB = (__bf16*)(ws + WCAT_OFF);
    __bf16* W6B   = (__bf16*)(ws + W6B_OFF);
    __bf16* W7B   = (__bf16*)(ws + W7B_OFF);
    float*  bcat  = (float*) (ws + BCAT_OFF);
    float*  gates = (float*) (ws + GATE_OFF);
    __bf16* hB    = (__bf16*)(ws + HB_OFF);
    float*  cbuf  = (float*) (ws + C_OFF);

    // prologue
    prep_kernel<<<(NN*HID + 255) / 256, 256, 0, stream>>>(
        W6, W7, W_ih, W_hh, b_ih, b_hh, W6B, W7B, WcatB, bcat, hB, cbuf);
    meanpool_kernel<<<(SEQ*BSZ*E) / 256, 256, 0, stream>>>(features, mpB);
    muconv_kernel<<<(ROWS*E) / 256, 256, 0, stream>>>(features, muB);
    xgemm_kernel<<<(ROWS/16) * 4 / 8, 256, 0, stream>>>(
        muB, mpB, W6B, W7B, b6, b7, features, xbuf);

    // sequential LSTM over S
    for (int s = 0; s < SEQ; ++s) {
        gates_gemm_kernel<<<(NN/16) * 16 / 8, 256, 0, stream>>>(
            xbuf, hB, WcatB, gates, s);
        lstm_update_kernel<<<NN / 8, 256, 0, stream>>>(
            gates, bcat, cbuf, hB, W5, b5, features, out, s);
    }
}
